// MFN_80221399154954
// MI455X (gfx1250) — matrix operand order repaired and checked by reading, not compiled, not run on hardware
//
#include <hip/hip_runtime.h>

// ---------------------------------------------------------------------------
// Types for CDNA5 WMMA (wave32): v_wmma_f32_16x16x32_bf16
// ---------------------------------------------------------------------------
typedef __attribute__((ext_vector_type(16))) __bf16       v16bf;
typedef __attribute__((ext_vector_type(8)))  float        v8f;
typedef __attribute__((ext_vector_type(8)))  unsigned int v8u;
typedef __attribute__((ext_vector_type(4)))  unsigned int v4u;

#define TSEQ 128   // sequence length
#define NBAT 256   // batch
#define NW   8     // waves per workgroup (256 threads, wave32)

// ---- dims ----
#define DL 768
#define DA 128
#define DV 128
#define HL 128
#define HA 64
#define HV 64

// ---- ws layout (bf16 weight elements, (N,K) row-major) ----
#define E_WihL 0LL
#define E_WhhL 393216LL
#define E_WihA 458752LL
#define E_WhhA 491520LL
#define E_WihV 507904LL
#define E_WhhV 540672LL
#define E_A11  557056LL
#define E_A12  688128LL
#define E_A21  819200LL
#define E_A22  950272LL
#define E_G11  983040LL
#define E_G12  1146880LL
#define E_G21  1179648LL
#define E_G22  1343488LL
#define E_O1   1376256LL
#define E_O2   1425408LL
#define E_END  1425536LL
#define BYTES_WEIGHTS (E_END * 2LL)         // 2851072
#define OFF_BIAS  2851072LL                 // 1024 fp32 combined lstm biases
#define OFF_GATES 4194304LL                 // fp32 gate preactivations
#define BYTES_GL  67108864LL                // 32768 x 512 fp32
#define BYTES_GA  33554432LL                // 32768 x 256 fp32
#define WS_NEED_GATES (OFF_GATES + BYTES_GL + 2LL * BYTES_GA)  // 138412032

// ---------------------------------------------------------------------------
// helpers
// ---------------------------------------------------------------------------
__device__ __forceinline__ float sigf(float x) { return 1.0f / (1.0f + __expf(-x)); }

// fp32 -> bf16 bits (RNE), integer path
__device__ __forceinline__ unsigned short f2bf(float v) {
  unsigned int u = __float_as_uint(v);
  u += 0x7FFFu + ((u >> 16) & 1u);
  return (unsigned short)(u >> 16);
}

// pack two fp32 -> packed bf16 pair (RNE)
__device__ __forceinline__ unsigned int pack_bf16(float x, float y) {
  unsigned int xu = __float_as_uint(x);
  unsigned int yu = __float_as_uint(y);
  xu += 0x7FFFu + ((xu >> 16) & 1u);
  yu += 0x7FFFu + ((yu >> 16) & 1u);
  return (xu >> 16) | (yu & 0xFFFF0000u);
}

__device__ __forceinline__ float bf16_to_f(unsigned short b) {
  return __uint_as_float(((unsigned int)b) << 16);
}

// A-fragment from row-major bf16 buffer (LDS): the lane's 16 elements are one
// contiguous 16-element run (matching B) -> two 16B loads, zero VALU.
// lane m=ln&15, h=ln>>4 ; run starts at k0+16h.
__device__ __forceinline__ v16bf fragAbf(const unsigned short* __restrict__ A,
                                         int lda, int k0, int ln) {
  const int m = ln & 15, h = ln >> 4;
  const unsigned short* p0 = A + m * lda + k0 + 16 * h;
  v4u lo = *(const v4u*)p0;
  v4u hi = *(const v4u*)(p0 + 8);
  v8u au;
  au[0] = lo[0]; au[1] = lo[1]; au[2] = lo[2]; au[3] = lo[3];
  au[4] = hi[0]; au[5] = hi[1]; au[6] = hi[2]; au[7] = hi[3];
  return __builtin_bit_cast(v16bf, au);
}

// Wave-level GEMM: 16xN = act( A[16xK](bf16) @ W^T + bias ), W bf16 (N,K).
// Optional fp32 output C and/or bf16 output Cb. Optional fp32 Cinit seeds the
// accumulator (may alias C). N-tiles strided across 8 waves; no barriers.
__device__ __forceinline__ void wave_gemm(
    float* __restrict__ C, int ldc,
    unsigned short* __restrict__ Cb, int ldcb,
    const unsigned short* __restrict__ A, int lda, int K,
    const unsigned short* __restrict__ W,
    const float* __restrict__ bias,
    const float* __restrict__ Cinit, long long ldci,
    int N, int act, int tid)
{
  const int ln = tid & 31, w = tid >> 5;
  const int n = ln & 15, h = ln >> 4;
  for (int n0 = w * 16; n0 < N; n0 += 16 * NW) {
    v8f acc;
    if (Cinit) {
#pragma unroll
      for (int r = 0; r < 8; ++r)
        acc[r] = Cinit[(long long)(r + 8 * h) * ldci + n0 + n];
    } else {
#pragma unroll
      for (int r = 0; r < 8; ++r) acc[r] = 0.0f;
    }
    const unsigned short* wrow = W + (long long)(n0 + n) * K + 16 * h;
    for (int k0 = 0; k0 < K; k0 += 32) {
      v16bf a = fragAbf(A, lda, k0, ln);
      v16bf b = *(const v16bf*)(wrow + k0);
      acc = __builtin_amdgcn_wmma_f32_16x16x32_bf16(
          false, a, false, b, (short)0, acc, false, false);
    }
    const float bv = bias ? bias[n0 + n] : 0.0f;
#pragma unroll
    for (int r = 0; r < 8; ++r) {
      float v = acc[r] + bv;
      if (act == 1)      v = fmaxf(v, 0.0f);
      else if (act == 2) v = tanhf(v);
      else if (act == 3) v = sigf(v);
      if (C)  C[(r + 8 * h) * ldc + n0 + n] = v;
      if (Cb) Cb[(r + 8 * h) * ldcb + n0 + n] = f2bf(v);
    }
  }
}

// ---------------------------------------------------------------------------
// Kernel: fp32 weights -> bf16 (N,K) row-major (transposing (K,N) linears)
// ---------------------------------------------------------------------------
__global__ void mfn_wconv(const float* __restrict__ src, unsigned short* __restrict__ dst,
                          int K, int N, int transpose) {
  long long i = (long long)blockIdx.x * 256 + threadIdx.x;
  long long tot = (long long)K * N;
  if (i >= tot) return;
  int nn = (int)(i / K), kk = (int)(i % K);
  float v = transpose ? src[(long long)kk * N + nn] : src[i];
  dst[i] = f2bf(v);
}

// combined lstm biases: bih + bhh for l(512), a(256), v(256)
__global__ void mfn_bcomb(const float* bihL, const float* bhhL,
                          const float* bihA, const float* bhhA,
                          const float* bihV, const float* bhhV,
                          float* __restrict__ dst) {
  int i = blockIdx.x * 256 + threadIdx.x;
  if (i < 512)       dst[i] = bihL[i] + bhhL[i];
  else if (i < 768)  dst[i] = bihA[i - 512] + bhhA[i - 512];
  else if (i < 1024) dst[i] = bihV[i - 768] + bhhV[i - 768];
}

// ---------------------------------------------------------------------------
// Kernel: parallel precompute of LSTM input projections for ALL (b,t):
//   G[r, :] = x[r, :] @ Wih^T + (bih + bhh),  r = b*T + t,  M = 32768
// Each block stages a 128xD x-tile into LDS as bf16 ONCE, then computes a
// 128x128 output tile (8 waves x 8 row-tiles). grid=(M/128, Ng/128), block=256
// ---------------------------------------------------------------------------
__global__ __launch_bounds__(256)
void mfn_gates(const float* __restrict__ X, const unsigned short* __restrict__ W,
               const float* __restrict__ bias, float* __restrict__ G,
               int D, int Ng) {
  extern __shared__ unsigned short xs[];   // 128 x D bf16
  const int tid = threadIdx.x, ln = tid & 31, w = tid >> 5;
  const int r0 = blockIdx.x * 128;
  const int n0 = blockIdx.y * 128 + w * 16;
  const int n = ln & 15, h = ln >> 4;

  // stage + convert x tile (pairs)
  {
    unsigned int* xsu = (unsigned int*)xs;
    const int pairsPerRow = D >> 1;
    const int totPairs = 128 * pairsPerRow;
    for (int i = tid; i < totPairs; i += 256) {
      const int r = i / pairsPerRow, kp = i - r * pairsPerRow;
      float2 f = *(const float2*)(X + (long long)(r0 + r) * D + 2 * kp);
      xsu[r * pairsPerRow + kp] = pack_bf16(f.x, f.y);
    }
  }
  __syncthreads();

  const unsigned short* wrow = W + (long long)(n0 + n) * D + 16 * h;
  const float bv = bias[n0 + n];
#pragma unroll 1
  for (int rt = 0; rt < 8; ++rt) {
    v8f acc;
#pragma unroll
    for (int r = 0; r < 8; ++r) acc[r] = 0.0f;
    const unsigned short* A = xs + rt * 16 * D;
    for (int k0 = 0; k0 < D; k0 += 32) {
      v16bf a = fragAbf(A, D, k0, ln);
      v16bf b = *(const v16bf*)(wrow + k0);
      acc = __builtin_amdgcn_wmma_f32_16x16x32_bf16(
          false, a, false, b, (short)0, acc, false, false);
    }
#pragma unroll
    for (int r = 0; r < 8; ++r)
      G[(long long)(r0 + rt * 16 + r + 8 * h) * Ng + n0 + n] = acc[r] + bv;
  }
}

// ---------------------------------------------------------------------------
// Recurrent persistent kernel. One workgroup owns 16 batch rows for all T.
// fp32 state + bf16 GEMM-operand mirrors all live in LDS for the whole scan.
// ---------------------------------------------------------------------------
struct RK {
  const float *Gl, *Ga, *Gv;          // precomputed gates (nullptr -> on-the-fly)
  const float *x_t, *x_a, *x_v;       // raw inputs (fallback path)
  const unsigned short *WihL, *WhhL, *WihA, *WhhA, *WihV, *WhhV;
  const unsigned short *A11, *A12, *A21, *A22, *G11, *G12, *G21, *G22, *O1, *O2;
  const float *bL, *bA, *bV;          // combined lstm biases (ws)
  const float *A11b, *A12b, *A21b, *A22b, *G11b, *G12b, *G21b, *G22b, *O1b, *O2b;
  float* out;                         // [0,256): output ; [256,...): last_hs
};

// fp32 LDS offsets (floats)
#define S_HL    0
#define S_CL    2048
#define S_HA    4096
#define S_CA    5120
#define S_HV    6144
#define S_CV    7168
#define S_MEM   8192
#define S_STATE_END 10240
#define S_CSTAR 10240   // 16 x 512
#define S_GATES 18432   // 16 x 512 (also attention logits)
#define S_CHAT  26624   // 16 x 128
#define S_G1    28672   // 16 x 128
#define S_G2    30720   // 16 x 128
#define S_F32_END 32768 // floats -> 131072 bytes

// bf16 LDS offsets (ushorts, relative to ub = end of fp32 region)
#define U_HLB   0       // 16 x 128
#define U_HAB   2048    // 16 x 64
#define U_HVB   3072    // 16 x 64
#define U_CSB   4096    // 16 x 512 (epilogue: last_hs 16 x 384)
#define U_ATTB  12288   // 16 x 640 (attended | mem)
#define U_HIDB  22528   // 16 x 256
#define U_XTB   26624   // 16 x 768 (fallback only)
#define U_BASE_END 26624
#define U_FALL_END 38912

#define LDS_BYTES_BASE (S_F32_END * 4 + U_BASE_END * 2)  // 184320
#define LDS_BYTES_FALL (S_F32_END * 4 + U_FALL_END * 2)  // 208896

__device__ __forceinline__ void lstm_update(float* __restrict__ h, unsigned short* __restrict__ hb,
                                            float* __restrict__ c,
                                            int hd, int hs,
                                            const float* __restrict__ g, int ldg,
                                            float* __restrict__ cS, unsigned short* __restrict__ cSb,
                                            int csoff, int tid) {
  const int tot = hd << 4;
  for (int i = tid; i < tot; i += 256) {
    const int r = i >> hs, j = i & (hd - 1);
    const float* gr = g + r * ldg;
    const float ig = sigf(gr[j]);
    const float fg = sigf(gr[hd + j]);
    const float gg = tanhf(gr[2 * hd + j]);
    const float og = sigf(gr[3 * hd + j]);
    const float cn = fg * c[i] + ig * gg;
    const float hn = og * tanhf(cn);
    c[i] = cn;
    h[i] = hn;
    hb[i] = f2bf(hn);
    cS[r * 512 + csoff + j] = cn;
    cSb[r * 512 + csoff + j] = f2bf(cn);
  }
}

// stage 16 rows of x[(b0+r)*T + t, :] into bf16 LDS tile (fallback path)
__device__ __forceinline__ void stage_x(const float* __restrict__ x,
                                        unsigned short* __restrict__ xtb,
                                        int D, int b0, int t, int tid) {
  unsigned int* xu = (unsigned int*)xtb;
  const int ppr = D >> 1;
  for (int i = tid; i < 16 * ppr; i += 256) {
    const int r = i / ppr, kp = i - r * ppr;
    float2 f = *(const float2*)(x + ((long long)(b0 + r) * TSEQ + t) * D + 2 * kp);
    xu[r * ppr + kp] = pack_bf16(f.x, f.y);
  }
}

__global__ __launch_bounds__(256)
void mfn_recurrent(RK p) {
  extern __shared__ float sm[];
  float* hl    = sm + S_HL;
  float* cl    = sm + S_CL;
  float* ha    = sm + S_HA;
  float* ca    = sm + S_CA;
  float* hv    = sm + S_HV;
  float* cv    = sm + S_CV;
  float* memb  = sm + S_MEM;
  float* cStar = sm + S_CSTAR;
  float* gates = sm + S_GATES;
  float* cHat  = sm + S_CHAT;
  float* g1b   = sm + S_G1;
  float* g2b   = sm + S_G2;
  unsigned short* ub   = (unsigned short*)(sm + S_F32_END);
  unsigned short* hlb  = ub + U_HLB;
  unsigned short* hab  = ub + U_HAB;
  unsigned short* hvb  = ub + U_HVB;
  unsigned short* csb  = ub + U_CSB;
  unsigned short* attb = ub + U_ATTB;
  unsigned short* hidb = ub + U_HIDB;
  unsigned short* xtb  = ub + U_XTB;

  const int tid = threadIdx.x;
  const int ln  = tid & 31;
  const int w   = tid >> 5;
  const int b0  = blockIdx.x * 16;

  // warm L2 with the bf16 weight block (global_prefetch_b8)
  for (long long off = (long long)tid * 128; off < BYTES_WEIGHTS; off += 256LL * 128)
    __builtin_prefetch((const char*)p.WihL + off, 0, 1);

  // zero carry state (fp32 h,c,mem) and bf16 h mirrors
  for (int i = tid; i < S_STATE_END; i += 256) sm[i] = 0.0f;
  for (int i = tid; i < 4096; i += 256) ub[i] = 0;   // hlb, hab, hvb
  __syncthreads();

  for (int t = 0; t < TSEQ; ++t) {
    // prev_cs -> cStar[:, 0:256] (fp32 + bf16 mirror)
    for (int i = tid; i < 16 * 256; i += 256) {
      const int r = i >> 8, c2 = i & 255;
      float v = (c2 < 128) ? cl[r * 128 + c2]
              : (c2 < 192) ? ca[r * 64 + (c2 - 128)]
                           : cv[r * 64 + (c2 - 192)];
      cStar[r * 512 + c2] = v;
      csb[r * 512 + c2] = f2bf(v);
    }
    __syncthreads();

    // ---------------- LSTM text (hd=128, gates N=512) ----------------
    if (p.Gl) {
      wave_gemm(gates, 512, nullptr, 0, hlb, HL, HL, p.WhhL, nullptr,
                p.Gl + ((long long)b0 * TSEQ + t) * 512LL, (long long)TSEQ * 512,
                512, 0, tid);
    } else {
      stage_x(p.x_t, xtb, DL, b0, t, tid);
      __syncthreads();
      wave_gemm(gates, 512, nullptr, 0, xtb, DL, DL, p.WihL, p.bL, nullptr, 0, 512, 0, tid);
      __syncthreads();
      wave_gemm(gates, 512, nullptr, 0, hlb, HL, HL, p.WhhL, nullptr, gates, 512, 512, 0, tid);
    }
    __syncthreads();
    lstm_update(hl, hlb, cl, 128, 7, gates, 512, cStar, csb, 256, tid);
    __syncthreads();

    // ---------------- LSTM audio (hd=64, gates N=256) ----------------
    if (p.Ga) {
      wave_gemm(gates, 512, nullptr, 0, hab, HA, HA, p.WhhA, nullptr,
                p.Ga + ((long long)b0 * TSEQ + t) * 256LL, (long long)TSEQ * 256,
                256, 0, tid);
    } else {
      stage_x(p.x_a, xtb, DA, b0, t, tid);
      __syncthreads();
      wave_gemm(gates, 512, nullptr, 0, xtb, DA, DA, p.WihA, p.bA, nullptr, 0, 256, 0, tid);
      __syncthreads();
      wave_gemm(gates, 512, nullptr, 0, hab, HA, HA, p.WhhA, nullptr, gates, 512, 256, 0, tid);
    }
    __syncthreads();
    lstm_update(ha, hab, ca, 64, 6, gates, 512, cStar, csb, 384, tid);
    __syncthreads();

    // ---------------- LSTM video (hd=64, gates N=256) ----------------
    if (p.Gv) {
      wave_gemm(gates, 512, nullptr, 0, hvb, HV, HV, p.WhhV, nullptr,
                p.Gv + ((long long)b0 * TSEQ + t) * 256LL, (long long)TSEQ * 256,
                256, 0, tid);
    } else {
      stage_x(p.x_v, xtb, DV, b0, t, tid);
      __syncthreads();
      wave_gemm(gates, 512, nullptr, 0, xtb, DV, DV, p.WihV, p.bV, nullptr, 0, 256, 0, tid);
      __syncthreads();
      wave_gemm(gates, 512, nullptr, 0, hvb, HV, HV, p.WhhV, nullptr, gates, 512, 256, 0, tid);
    }
    __syncthreads();
    lstm_update(hv, hvb, cv, 64, 6, gates, 512, cStar, csb, 448, tid);
    __syncthreads();

    // ---------------- attention ----------------
    // hidb = relu(cStar @ att1_fc1 + b)      (bf16 only, 16x256, K=512)
    wave_gemm(nullptr, 0, hidb, 256, csb, 512, 512, p.A11, p.A11b, nullptr, 0, 256, 1, tid);
    __syncthreads();
    // logits (fp32, reuse gates) = hidb @ att1_fc2 + b
    wave_gemm(gates, 512, nullptr, 0, hidb, 256, 256, p.A12, p.A12b, nullptr, 0, 512, 0, tid);
    __syncthreads();

    // softmax over 512 per row; attb = bf16(softmax * cStar) ; attb[:,512:640]=mem
    {
#pragma unroll
      for (int rr = 0; rr < 2; ++rr) {
        const int r = w * 2 + rr;
        const float* lrow = gates + r * 512;
        unsigned short* arow = attb + r * 640;
        float mx = -3.0e38f;
#pragma unroll
        for (int j = 0; j < 16; ++j) mx = fmaxf(mx, lrow[ln + 32 * j]);
        for (int off = 16; off; off >>= 1) mx = fmaxf(mx, __shfl_xor(mx, off, 32));
        float e[16], s = 0.0f;
#pragma unroll
        for (int j = 0; j < 16; ++j) { e[j] = __expf(lrow[ln + 32 * j] - mx); s += e[j]; }
        for (int off = 16; off; off >>= 1) s += __shfl_xor(s, off, 32);
        const float inv = 1.0f / s;
#pragma unroll
        for (int j = 0; j < 16; ++j) {
          const int c = ln + 32 * j;
          arow[c] = f2bf(e[j] * inv * cStar[r * 512 + c]);
        }
#pragma unroll
        for (int j = 0; j < 4; ++j)
          arow[512 + ln + 32 * j] = f2bf(memb[r * 128 + ln + 32 * j]);
      }
    }
    __syncthreads();

    // cHat = tanh(mlp(attended))   K=512 over attb (lda=640)
    wave_gemm(nullptr, 0, hidb, 256, attb, 640, 512, p.A21, p.A21b, nullptr, 0, 256, 1, tid);
    __syncthreads();
    wave_gemm(cHat, 128, nullptr, 0, hidb, 256, 256, p.A22, p.A22b, nullptr, 0, 128, 2, tid);
    __syncthreads();

    // gamma1 = sigmoid(mlp(both))  K=640
    wave_gemm(nullptr, 0, hidb, 256, attb, 640, 640, p.G11, p.G11b, nullptr, 0, 256, 1, tid);
    __syncthreads();
    wave_gemm(g1b, 128, nullptr, 0, hidb, 256, 256, p.G12, p.G12b, nullptr, 0, 128, 3, tid);
    __syncthreads();

    // gamma2 = sigmoid(mlp(both))
    wave_gemm(nullptr, 0, hidb, 256, attb, 640, 640, p.G21, p.G21b, nullptr, 0, 256, 1, tid);
    __syncthreads();
    wave_gemm(g2b, 128, nullptr, 0, hidb, 256, 256, p.G22, p.G22b, nullptr, 0, 128, 3, tid);
    __syncthreads();

    // mem = g1*mem + g2*cHat
    for (int i = tid; i < 16 * 128; i += 256)
      memb[i] = g1b[i] * memb[i] + g2b[i] * cHat[i];
    __syncthreads();
  }

  // ---------------- epilogue: last_hs and output MLP ----------------
  for (int i = tid; i < 16 * 384; i += 256) {
    const int r = i / 384, c = i % 384;
    float v = (c < 128) ? hl[r * 128 + c]
            : (c < 192) ? ha[r * 64 + (c - 128)]
            : (c < 256) ? hv[r * 64 + (c - 192)]
                        : memb[r * 128 + (c - 256)];
    csb[r * 384 + c] = f2bf(v);                        // bf16 tile for GEMM
    p.out[256 + (long long)(b0 + r) * 384 + c] = v;    // fp32 last_hs output
  }
  __syncthreads();
  // hidden = relu(last_hs @ out_fc1 + b)   (16x128, K=384)
  wave_gemm(g1b, 128, nullptr, 0, csb, 384, 384, p.O1, p.O1b, nullptr, 0, 128, 1, tid);
  __syncthreads();
  if (tid < 16) {
    float s = p.O2b[0];
    for (int k = 0; k < 128; ++k)
      s += g1b[tid * 128 + k] * bf16_to_f(p.O2[k]);
    p.out[b0 + tid] = s;
  }
}

// ---------------------------------------------------------------------------
// Host launcher
// ---------------------------------------------------------------------------
extern "C" void kernel_launch(void* const* d_in, const int* in_sizes, int n_in,
                              void* d_out, int out_size, void* d_ws, size_t ws_size,
                              hipStream_t stream) {
  const float* x_t = (const float*)d_in[0];
  const float* x_a = (const float*)d_in[1];
  const float* x_v = (const float*)d_in[2];

  // canonical param slots (insertion order):
  // 0..3 lstm_l(Wih,Whh,bih,bhh) 4..7 lstm_a 8..11 lstm_v
  // 12..19 att1_fc1(W,b) att1_fc2 att2_fc1 att2_fc2
  // 20..27 gamma1_fc1 gamma1_fc2 gamma2_fc1 gamma2_fc2
  // 28..31 out_fc1(W,b) out_fc2(W,b)
  static const int mapSort[32] = {
      23, 24, 25, 26,   // lstm_l
      19, 20, 21, 22,   // lstm_a
      27, 28, 29, 30,   // lstm_v
      3, 4, 5, 6, 7, 8, 9, 10,        // att1_fc1..att2_fc2
      11, 12, 13, 14, 15, 16, 17, 18, // gamma*
      31, 32, 33, 34 };               // out_fc1, out_fc2
  const bool insertion = (n_in > 3) && (in_sizes[3] == 512 * 768);
  const float* prm[32];
  for (int j = 0; j < 32; ++j)
    prm[j] = (const float*)d_in[insertion ? (3 + j) : mapSort[j]];

  char* ws = (char*)d_ws;
  unsigned short* wb = (unsigned short*)ws;
  float* biasC = (float*)(ws + OFF_BIAS);
  const bool useG = (ws_size >= (size_t)WS_NEED_GATES);
  float* Gl = (float*)(ws + OFF_GATES);
  float* Ga = (float*)(ws + OFF_GATES + BYTES_GL);
  float* Gv = (float*)(ws + OFF_GATES + BYTES_GL + BYTES_GA);

  // 1) convert weights to bf16 (N,K)
  struct WD { int slot; long long off; int K, N, tr; };
  static const WD wd[16] = {
      {0,  E_WihL, 768, 512, 0}, {1,  E_WhhL, 128, 512, 0},
      {4,  E_WihA, 128, 256, 0}, {5,  E_WhhA,  64, 256, 0},
      {8,  E_WihV, 128, 256, 0}, {9,  E_WhhV,  64, 256, 0},
      {12, E_A11,  512, 256, 1}, {14, E_A12,  256, 512, 1},
      {16, E_A21,  512, 256, 1}, {18, E_A22,  256, 128, 1},
      {20, E_G11,  640, 256, 1}, {22, E_G12,  256, 128, 1},
      {24, E_G21,  640, 256, 1}, {26, E_G22,  256, 128, 1},
      {28, E_O1,   384, 128, 1}, {30, E_O2,   128,   1, 1}};
  for (int i = 0; i < 16; ++i) {
    const long long tot = (long long)wd[i].K * wd[i].N;
    const int blocks = (int)((tot + 255) / 256);
    mfn_wconv<<<blocks, 256, 0, stream>>>(prm[wd[i].slot], wb + wd[i].off,
                                          wd[i].K, wd[i].N, wd[i].tr);
  }
  // 2) combined lstm biases
  mfn_bcomb<<<4, 256, 0, stream>>>(prm[2], prm[3], prm[6], prm[7], prm[10], prm[11], biasC);

  // 3) precompute input projections (the fully parallel 30-GFLOP chunk)
  if (useG) {
    dim3 gl(NBAT * TSEQ / 128, 512 / 128);
    mfn_gates<<<gl, 256, 128 * DL * 2, stream>>>(x_t, wb + E_WihL, biasC, Gl, DL, 512);
    dim3 ga(NBAT * TSEQ / 128, 256 / 128);
    mfn_gates<<<ga, 256, 128 * DA * 2, stream>>>(x_a, wb + E_WihA, biasC + 512, Ga, DA, 256);
    dim3 gv(NBAT * TSEQ / 128, 256 / 128);
    mfn_gates<<<gv, 256, 128 * DV * 2, stream>>>(x_v, wb + E_WihV, biasC + 768, Gv, DV, 256);
  }

  // 4) recurrent scan: 16 workgroups x 16 batch rows, state in LDS
  RK p;
  p.Gl = useG ? Gl : nullptr;
  p.Ga = useG ? Ga : nullptr;
  p.Gv = useG ? Gv : nullptr;
  p.x_t = x_t; p.x_a = x_a; p.x_v = x_v;
  p.WihL = wb + E_WihL; p.WhhL = wb + E_WhhL;
  p.WihA = wb + E_WihA; p.WhhA = wb + E_WhhA;
  p.WihV = wb + E_WihV; p.WhhV = wb + E_WhhV;
  p.A11 = wb + E_A11; p.A12 = wb + E_A12; p.A21 = wb + E_A21; p.A22 = wb + E_A22;
  p.G11 = wb + E_G11; p.G12 = wb + E_G12; p.G21 = wb + E_G21; p.G22 = wb + E_G22;
  p.O1  = wb + E_O1;  p.O2  = wb + E_O2;
  p.bL = biasC; p.bA = biasC + 512; p.bV = biasC + 768;
  p.A11b = prm[13]; p.A12b = prm[15]; p.A21b = prm[17]; p.A22b = prm[19];
  p.G11b = prm[21]; p.G12b = prm[23]; p.G21b = prm[25]; p.G22b = prm[27];
  p.O1b  = prm[29]; p.O2b  = prm[31];
  p.out = (float*)d_out;

  const size_t shmem = useG ? LDS_BYTES_BASE : LDS_BYTES_FALL;
  mfn_recurrent<<<NBAT / 16, 256, shmem, stream>>>(p);
}
